// mamba_20590073217681
// MI455X (gfx1250) — matrix operand order repaired and checked by reading, not compiled, not run on hardware
//
#include <hip/hip_runtime.h>
#include <math.h>

#define DMODEL 96
#define DINNER 96
#define NSTATE 64
#define DTRANK 6
#define KDIR   4
#define BATCH  4
#define SEQ    16384
#define BL     (BATCH*SEQ)         // 65536 rows
#define CHUNK  256
#define NC     (SEQ/CHUNK)         // 64 chunks
#define NSUB   (CHUNK/16)          // 16 subtiles per chunk
#define PROJC  (DINNER + 2*NSTATE) // 224 fused projection columns
#define LOG2E  1.4426950408889634f
#define LN2    0.6931471805599453f

typedef __attribute__((ext_vector_type(16))) __bf16 v16bf;
typedef __attribute__((ext_vector_type(8)))  float  v8f;
typedef __attribute__((ext_vector_type(4)))  unsigned v4u;
typedef __attribute__((ext_vector_type(8)))  int      v8i;
typedef __attribute__((ext_vector_type(4)))  int      v4i;

// ---- fast transcendental helpers (raw HW ops, no library slow paths) ------

__device__ __forceinline__ float fexp2(float x) {
#if __has_builtin(__builtin_amdgcn_exp2f)
  return __builtin_amdgcn_exp2f(x);
#else
  return exp2f(x);
#endif
}
__device__ __forceinline__ float flog2(float x) {
#if __has_builtin(__builtin_amdgcn_logf)
  return __builtin_amdgcn_logf(x);
#else
  return log2f(x);
#endif
}
__device__ __forceinline__ float frcp(float x) {
#if __has_builtin(__builtin_amdgcn_rcpf)
  return __builtin_amdgcn_rcpf(x);
#else
  return 1.f / x;
#endif
}
__device__ __forceinline__ float fexp(float x) { return fexp2(x * LOG2E); }
// branchless, numerically stable softplus
__device__ __forceinline__ float softplusf(float x) {
  return fmaxf(x, 0.f) + LN2 * flog2(1.f + fexp2(-fabsf(x) * LOG2E));
}
// branchless erf (Abramowitz-Stegun 7.1.26, |err| <= 1.5e-7), native rcp/exp
__device__ __forceinline__ float erf_fast(float x) {
  float ax = fabsf(x);
  float t = frcp(fmaf(0.3275911f, ax, 1.f));
  float p = fmaf(t, 1.061405429f, -1.453152027f);
  p = fmaf(t, p, 1.421413741f);
  p = fmaf(t, p, -0.284496736f);
  p = fmaf(t, p, 0.254829592f);
  p = p * t;
  float e = fexp2(-ax * ax * LOG2E);
  float r = fmaf(-p, e, 1.f);
  return copysignf(r, x);
}
__device__ __forceinline__ float geluf(float x) {
  return 0.5f * x * (1.f + erf_fast(x * 0.70710678118654752f));
}

// ---- WMMA fragment helpers -------------------------------------------------

__device__ __forceinline__ __bf16 f2bf(float f) { return (__bf16)f; }  // native cvt

// A-matrix 16x32 bf16 element->K map (contiguous, matches kmapB)
__device__ __forceinline__ int kmapA(int e, int half) {
  return e + half * 16;
}
// B-matrix 32x16 bf16 element->K map (contiguous, matches kmapA)
__device__ __forceinline__ int kmapB(int e, int half) {
  return e + half * 16;
}

// Build A fragment (16 rows x 32 k) from LDS f32 tile; optional row flip for
// reversed-direction tiles that were DMA'd in forward memory order.
__device__ __forceinline__ v16bf a_frag_lds(const float* t, int stride, int kbase,
                                            int lane, int flip) {
  int m = lane & 15, hf = lane >> 4;
  if (flip) m = 15 - m;
  v16bf a;
#pragma unroll
  for (int e = 0; e < 16; ++e) a[e] = f2bf(t[m * stride + kbase + kmapA(e, hf)]);
  return a;
}

__device__ __forceinline__ v8f wmma_bf16(v16bf a, v16bf b, v8f c) {
  return __builtin_amdgcn_wmma_f32_16x16x32_bf16(false, a, false, b, (short)0, c, false, false);
}

// Fused projection weight element: rows 0..95 = W_delta[k], rows 96..223 = x_proj_w[k] rows 6..133
__device__ __forceinline__ float wcat_elem(const float* wd_k, const float* xp_k, int c, int j) {
  return (c < DINNER) ? wd_k[c * DINNER + j] : xp_k[(c - 90) * DINNER + j];
}

__device__ __forceinline__ v16bf b_frag_wcat(const float* wd_k, const float* xp_k,
                                             int cbase, int kbase, int lane) {
  int n = lane & 15, hf = lane >> 4;
  v16bf b;
#pragma unroll
  for (int e = 0; e < 16; ++e) b[e] = f2bf(wcat_elem(wd_k, xp_k, cbase + n, kbase + kmapB(e, hf)));
  return b;
}

__device__ __forceinline__ v16bf b_frag_global(const float* w, int wstride,
                                               int cbase, int kbase, int lane) {
  int n = lane & 15, hf = lane >> 4;
  v16bf b;
#pragma unroll
  for (int e = 0; e < 16; ++e) b[e] = f2bf(w[(cbase + n) * wstride + kbase + kmapB(e, hf)]);
  return b;
}

// ---- Tensor Data Mover: load one 16x96 f32 tile (row-major, stride 96) to LDS

__device__ __forceinline__ void tdm_load_tile16x96(unsigned lds_off, const float* gptr) {
  unsigned long long ga = (unsigned long long)(uintptr_t)gptr;
  v4u g0;
  g0[0] = 1u;                                              // count=1, user mode
  g0[1] = lds_off;                                         // lds_addr (bytes)
  g0[2] = (unsigned)(ga & 0xffffffffu);                    // global_addr[31:0]
  g0[3] = (unsigned)((ga >> 32) & 0x01ffffffu) | (2u << 30); // global_addr[56:32] | type=2
  v8i g1;
  g1[0] = (int)(2u << 16);        // workgroup_mask=0, data_size=2 (4 bytes)
  g1[1] = (int)(96u << 16);       // atomic_addr=0 | tensor_dim0[15:0]=96
  g1[2] = (int)(16u << 16);       // tensor_dim0[31:16]=0 | tensor_dim1[15:0]=16
  g1[3] = (int)(96u << 16);       // tensor_dim1[31:16]=0 | tile_dim0=96
  g1[4] = (int)16;                // tile_dim1=16 | tile_dim2=0
  g1[5] = (int)96;                // tensor_dim0_stride[31:0]=96
  g1[6] = 0;                      // stride0[47:32]=0 | stride1[15:0]=0
  g1[7] = 0;
  v4i z4 = {0, 0, 0, 0};
#if __has_include(<hip/amd_detail/amd_gfx1250_TDM.h>)
  v8i z8 = {0, 0, 0, 0, 0, 0, 0, 0};
  __builtin_amdgcn_tensor_load_to_lds(g0, g1, z4, z4, z8, 0);
#else
  __builtin_amdgcn_tensor_load_to_lds(g0, g1, z4, z4, 0);
#endif
}

__device__ __forceinline__ unsigned lds_off_of(const void* p) {
  return (unsigned)(uintptr_t)p;  // low 32 bits of generic addr = LDS byte offset
}

// ---- kernel 0: W_delta[k] = dt_projs_w[k] (96x6) @ x_proj_w[k][0:6] (6x96) --

__global__ void wdelta_kernel(const float* __restrict__ dtw, const float* __restrict__ xpw,
                              float* __restrict__ wdelta) {
  int k = blockIdx.x;
  for (int i = threadIdx.x; i < DINNER * DINNER; i += blockDim.x) {
    int d = i / DINNER, j = i % DINNER;
    float acc = 0.f;
#pragma unroll
    for (int r = 0; r < DTRANK; ++r)
      acc += dtw[(k * DINNER + d) * DTRANK + r] * xpw[(k * PROJC + r) * DINNER + j];
    wdelta[k * DINNER * DINNER + i] = acc;
  }
}

// ---- kernel 1: in_proj (WMMA, TDM-staged A tiles) -> xx_raw, z=gelu -------

__global__ void __launch_bounds__(128) inproj_kernel(const float* __restrict__ x,
                                                     const float* __restrict__ w,
                                                     float* __restrict__ xxraw,
                                                     float* __restrict__ zbuf) {
  __shared__ __align__(16) float s_x[2][16 * DMODEL];
  int tid = threadIdx.x, lane = tid & 31, wv = tid >> 5;

  // 12 N-tiles of 16 over 4 waves: wave w owns tiles w, w+4, w+8
  v16bf bw[3][3];
#pragma unroll
  for (int i = 0; i < 3; ++i)
#pragma unroll
    for (int kc = 0; kc < 3; ++kc)
      bw[i][kc] = b_frag_global(w, DMODEL, (wv + 4 * i) * 16, kc * 32, lane);

  size_t rowbase = (size_t)blockIdx.x * 8 * 16;
  if (tid < 32) tdm_load_tile16x96(lds_off_of(s_x[0]), x + rowbase * DMODEL);

  for (int ms = 0; ms < 8; ++ms) {
    size_t row0 = rowbase + (size_t)ms * 16;
    if (tid < 32) {
      if (ms + 1 < 8) {
        tdm_load_tile16x96(lds_off_of(s_x[(ms + 1) & 1]), x + (row0 + 16) * DMODEL);
        __builtin_amdgcn_s_wait_tensorcnt(1);
      } else {
        __builtin_amdgcn_s_wait_tensorcnt(0);
      }
    }
    __syncthreads();
    const float* sx = s_x[ms & 1];
    v16bf af0 = a_frag_lds(sx, DMODEL, 0, lane, 0);
    v16bf af1 = a_frag_lds(sx, DMODEL, 32, lane, 0);
    v16bf af2 = a_frag_lds(sx, DMODEL, 64, lane, 0);
#pragma unroll
    for (int i = 0; i < 3; ++i) {
      int tile = wv + 4 * i;               // wave-uniform
      v8f acc = {};
      acc = wmma_bf16(af0, bw[i][0], acc);
      acc = wmma_bf16(af1, bw[i][1], acc);
      acc = wmma_bf16(af2, bw[i][2], acc);
      int n = lane & 15;
      int rb = (lane >> 4) * 8;
      if (tile < 6) {                      // xx region (c = tile*16+n < 96)
        int c = tile * 16 + n;
#pragma unroll
        for (int v = 0; v < 8; ++v) xxraw[(row0 + v + rb) * DINNER + c] = acc[v];
      } else {                             // z region -> gelu
        int c = (tile - 6) * 16 + n;
#pragma unroll
        for (int v = 0; v < 8; ++v) zbuf[(row0 + v + rb) * DINNER + c] = geluf(acc[v]);
      }
    }
    __syncthreads();
  }
}

// ---- kernel 2: depthwise conv1d (k=3, same) + gelu -> u -------------------

__global__ void conv_kernel(const float* __restrict__ xxraw, const float* __restrict__ cw,
                            float* __restrict__ u) {
  int idx = blockIdx.x * 256 + threadIdx.x;
  if (idx >= BL * DINNER) return;
  int d = idx % DINNER;
  int l = (idx / DINNER) % SEQ;
  int b = idx / (DINNER * SEQ);
  const float* base = xxraw + (size_t)b * SEQ * DINNER + d;
  if (l + 64 < SEQ) __builtin_prefetch(base + (size_t)(l + 64) * DINNER, 0, 1);
  float acc = cw[d * 3 + 1] * base[(size_t)l * DINNER];
  if (l > 0)       acc += cw[d * 3 + 0] * base[(size_t)(l - 1) * DINNER];
  if (l < SEQ - 1) acc += cw[d * 3 + 2] * base[(size_t)(l + 1) * DINNER];
  u[idx] = geluf(acc);
}

// ---- kernel 3: pass1 — per-chunk scan, WMMA projections, TDM staging ------
// block = 128 (4 waves); grid = B*K*NC. Thread d (<96) owns channel d with all
// 64 state columns in registers; no cross-lane reductions.

__global__ void __launch_bounds__(128) scan_pass1(
    const float* __restrict__ u, const float* __restrict__ xpw,
    const float* __restrict__ wdelta, const float* __restrict__ A_logs,
    const float* __restrict__ Ds, float* __restrict__ ys,
    float* __restrict__ hend, float* __restrict__ Ssum) {
  __shared__ __align__(16) float s_u[2][16 * DINNER];
  __shared__ __align__(16) float s_delta[16 * DINNER];
  __shared__ __align__(16) float s_B[16 * NSTATE];
  __shared__ __align__(16) float s_C[16 * NSTATE];
  __shared__ __align__(16) float s_y[16 * DINNER];

  int tid = threadIdx.x, lane = tid & 31, wv = tid >> 5;
  int bk = blockIdx.x / NC, chunk = blockIdx.x % NC;
  int b = bk >> 2, k = bk & 3;
  int flip = k & 1;
  const float* wd_k = wdelta + k * DINNER * DINNER;
  const float* xp_k = xpw + k * PROJC * DINNER;

  // 14 N-tiles over 4 waves: wave w owns tiles w, w+4, w+8, (w+12 if <14)
  v16bf bw[4][3];
#pragma unroll
  for (int i = 0; i < 4; ++i) {
    int tile = wv + 4 * i;
    if (tile < 14)
#pragma unroll
      for (int kc = 0; kc < 3; ++kc)
        bw[i][kc] = b_frag_wcat(wd_k, xp_k, tile * 16, kc * 32, lane);
  }

  // per-thread scan state: channel d = tid (threads 96..127 only help with GEMM/IO)
  int d = tid;
  float h[NSTATE], a2[NSTATE];
  float dsc = 0.f, ssum = 0.f;
  if (d < DINNER) {
    const float4* al4 = (const float4*)&A_logs[(k * DINNER + d) * NSTATE];
#pragma unroll
    for (int q = 0; q < 16; ++q) {
      float4 al = al4[q];
      a2[4 * q + 0] = -fexp(al.x) * LOG2E;
      a2[4 * q + 1] = -fexp(al.y) * LOG2E;
      a2[4 * q + 2] = -fexp(al.z) * LOG2E;
      a2[4 * q + 3] = -fexp(al.w) * LOG2E;
      h[4 * q + 0] = 0.f; h[4 * q + 1] = 0.f; h[4 * q + 2] = 0.f; h[4 * q + 3] = 0.f;
    }
    dsc = Ds[k * DINNER + d];
  }

  float* ysk = ys + (size_t)k * BL * DINNER;
  int tbase0 = chunk * CHUNK;

  auto tile_ptr = [&](int sub) -> const float* {
    int tb = tbase0 + sub * 16;
    int first = flip ? (SEQ - 1 - (tb + 15)) : tb;
    return u + ((size_t)b * SEQ + first) * DINNER;
  };

  if (tid < 32) tdm_load_tile16x96(lds_off_of(s_u[0]), tile_ptr(0));

  for (int sub = 0; sub < NSUB; ++sub) {
    int tb = tbase0 + sub * 16;
    if (tid < 32) {
      if (sub + 1 < NSUB) {
        tdm_load_tile16x96(lds_off_of(s_u[(sub + 1) & 1]), tile_ptr(sub + 1));
        __builtin_amdgcn_s_wait_tensorcnt(1);
      } else {
        __builtin_amdgcn_s_wait_tensorcnt(0);
      }
    }
    __syncthreads();
    const float* su = s_u[sub & 1];

    // projections: delta/B/C for 16 timesteps
    v16bf af0 = a_frag_lds(su, DINNER, 0, lane, flip);
    v16bf af1 = a_frag_lds(su, DINNER, 32, lane, flip);
    v16bf af2 = a_frag_lds(su, DINNER, 64, lane, flip);
#pragma unroll
    for (int i = 0; i < 4; ++i) {
      int tile = wv + 4 * i;               // wave-uniform
      if (tile < 14) {
        v8f acc = {};
        acc = wmma_bf16(af0, bw[i][0], acc);
        acc = wmma_bf16(af1, bw[i][1], acc);
        acc = wmma_bf16(af2, bw[i][2], acc);
        int n = lane & 15;
        int rb = (lane >> 4) * 8;
        if (tile < 6) {                    // delta region (uniform branch)
          int c = tile * 16 + n;
#pragma unroll
          for (int v = 0; v < 8; ++v) s_delta[(v + rb) * DINNER + c] = softplusf(acc[v]);
        } else if (tile < 10) {            // B region
          int c = (tile - 6) * 16 + n;
#pragma unroll
          for (int v = 0; v < 8; ++v) s_B[(v + rb) * NSTATE + c] = acc[v];
        } else {                           // C region
          int c = (tile - 10) * 16 + n;
#pragma unroll
          for (int v = 0; v < 8; ++v) s_C[(v + rb) * NSTATE + c] = acc[v];
        }
      }
    }
    __syncthreads();

    // scan 16 steps: thread per channel, state in registers, no shuffles
    if (d < DINNER) {
      for (int tt = 0; tt < 16; ++tt) {
        int ur = flip ? (15 - tt) : tt;
        float dl = s_delta[tt * DINNER + d];
        float ut = su[ur * DINNER + d];
        float du = dl * ut;
        const float4* B4 = (const float4*)&s_B[tt * NSTATE];
        const float4* C4 = (const float4*)&s_C[tt * NSTATE];
        float y0 = 0.f, y1 = 0.f, y2 = 0.f, y3 = 0.f;
#pragma unroll
        for (int q = 0; q < 16; ++q) {
          float4 bb = B4[q], cc = C4[q];
          h[4 * q + 0] = fexp2(dl * a2[4 * q + 0]) * h[4 * q + 0] + du * bb.x;
          h[4 * q + 1] = fexp2(dl * a2[4 * q + 1]) * h[4 * q + 1] + du * bb.y;
          h[4 * q + 2] = fexp2(dl * a2[4 * q + 2]) * h[4 * q + 2] + du * bb.z;
          h[4 * q + 3] = fexp2(dl * a2[4 * q + 3]) * h[4 * q + 3] + du * bb.w;
          y0 += h[4 * q + 0] * cc.x;
          y1 += h[4 * q + 1] * cc.y;
          y2 += h[4 * q + 2] * cc.z;
          y3 += h[4 * q + 3] * cc.w;
        }
        s_y[tt * DINNER + d] = (y0 + y1) + (y2 + y3) + dsc * ut;
        ssum += dl;
      }
    }
    __syncthreads();
    for (int i = tid; i < 16 * DINNER; i += 128) {
      int r = i / DINNER;
      ysk[((size_t)b * SEQ + (tb + r)) * DINNER + (i % DINNER)] = s_y[i];
    }
    __syncthreads();
  }

  // store per-chunk carries
  if (d < DINNER) {
    size_t cb = (size_t)bk * NC + chunk;
    float4* h4 = (float4*)&hend[(cb * DINNER + d) * NSTATE];
#pragma unroll
    for (int q = 0; q < 16; ++q)
      h4[q] = make_float4(h[4 * q + 0], h[4 * q + 1], h[4 * q + 2], h[4 * q + 3]);
    Ssum[cb * DINNER + d] = ssum;
  }
}

// ---- kernel 4: pass2 — combine chunk carries (in-place: hend -> h_in) -----

__global__ void scan_pass2(const float* __restrict__ A_logs, const float* __restrict__ Ssum,
                           float* __restrict__ hend) {
  int idx = blockIdx.x * 256 + threadIdx.x;   // over B*K*96*64
  int n = idx & 63;
  int dch = (idx >> 6) % DINNER;
  int bk = idx / (DINNER * NSTATE);
  int k = bk & 3;
  float a2 = -fexp(A_logs[(k * DINNER + dch) * NSTATE + n]) * LOG2E;
  float carry = 0.f;
  for (int c = 0; c < NC; ++c) {
    size_t cb = (size_t)bk * NC + c;
    size_t hidx = (cb * DINNER + dch) * NSTATE + n;
    float he = hend[hidx];
    float s  = Ssum[cb * DINNER + dch];
    hend[hidx] = carry;                       // becomes h_in for chunk c
    carry = he + fexp2(a2 * s) * carry;
  }
}

// ---- kernel 5: pass3 — carry-in correction y += C_t . (exp(A*cum)*h_in) ---

__global__ void __launch_bounds__(128) scan_pass3(
    const float* __restrict__ u, const float* __restrict__ xpw,
    const float* __restrict__ wdelta, const float* __restrict__ A_logs,
    const float* __restrict__ hin, float* __restrict__ ys) {
  __shared__ __align__(16) float s_u[2][16 * DINNER];
  __shared__ __align__(16) float s_delta[16 * DINNER];
  __shared__ __align__(16) float s_C[16 * NSTATE];
  __shared__ __align__(16) float s_y[16 * DINNER];

  int tid = threadIdx.x, lane = tid & 31, wv = tid >> 5;
  int bk = blockIdx.x / (NC - 1);
  int chunk = 1 + (blockIdx.x % (NC - 1));
  int b = bk >> 2, k = bk & 3;
  int flip = k & 1;
  const float* wd_k = wdelta + k * DINNER * DINNER;
  const float* xp_k = xpw + k * PROJC * DINNER;

  // needed tiles {0..5 (delta), 10..13 (C)} indexed i=0..9, tile = i<6 ? i : i+4
  v16bf bw[3][3];
#pragma unroll
  for (int j = 0; j < 3; ++j) {
    int i = wv + 4 * j;
    if (i < 10) {
      int tile = (i < 6) ? i : i + 4;
#pragma unroll
      for (int kc = 0; kc < 3; ++kc)
        bw[j][kc] = b_frag_wcat(wd_k, xp_k, tile * 16, kc * 32, lane);
    }
  }

  int d = tid;
  float hn[NSTATE], a2[NSTATE];
  float cum = 0.f;
  size_t cb = (size_t)bk * NC + chunk;
  if (d < DINNER) {
    const float4* al4 = (const float4*)&A_logs[(k * DINNER + d) * NSTATE];
    const float4* hi4 = (const float4*)&hin[(cb * DINNER + d) * NSTATE];
#pragma unroll
    for (int q = 0; q < 16; ++q) {
      float4 al = al4[q];
      float4 hv = hi4[q];
      a2[4 * q + 0] = -fexp(al.x) * LOG2E;
      a2[4 * q + 1] = -fexp(al.y) * LOG2E;
      a2[4 * q + 2] = -fexp(al.z) * LOG2E;
      a2[4 * q + 3] = -fexp(al.w) * LOG2E;
      hn[4 * q + 0] = hv.x; hn[4 * q + 1] = hv.y; hn[4 * q + 2] = hv.z; hn[4 * q + 3] = hv.w;
    }
  }

  float* ysk = ys + (size_t)k * BL * DINNER;
  int tbase0 = chunk * CHUNK;
  auto tile_ptr = [&](int sub) -> const float* {
    int tb = tbase0 + sub * 16;
    int first = flip ? (SEQ - 1 - (tb + 15)) : tb;
    return u + ((size_t)b * SEQ + first) * DINNER;
  };

  if (tid < 32) tdm_load_tile16x96(lds_off_of(s_u[0]), tile_ptr(0));

  for (int sub = 0; sub < NSUB; ++sub) {
    int tb = tbase0 + sub * 16;
    if (tid < 32) {
      if (sub + 1 < NSUB) {
        tdm_load_tile16x96(lds_off_of(s_u[(sub + 1) & 1]), tile_ptr(sub + 1));
        __builtin_amdgcn_s_wait_tensorcnt(1);
      } else {
        __builtin_amdgcn_s_wait_tensorcnt(0);
      }
    }
    __syncthreads();
    const float* su = s_u[sub & 1];

    v16bf af0 = a_frag_lds(su, DINNER, 0, lane, flip);
    v16bf af1 = a_frag_lds(su, DINNER, 32, lane, flip);
    v16bf af2 = a_frag_lds(su, DINNER, 64, lane, flip);
#pragma unroll
    for (int j = 0; j < 3; ++j) {
      int i = wv + 4 * j;
      if (i < 10) {
        int tile = (i < 6) ? i : i + 4;    // wave-uniform
        v8f acc = {};
        acc = wmma_bf16(af0, bw[j][0], acc);
        acc = wmma_bf16(af1, bw[j][1], acc);
        acc = wmma_bf16(af2, bw[j][2], acc);
        int n = lane & 15;
        int rb = (lane >> 4) * 8;
        if (tile < 6) {                    // delta region
          int c = tile * 16 + n;
#pragma unroll
          for (int v = 0; v < 8; ++v) s_delta[(v + rb) * DINNER + c] = softplusf(acc[v]);
        } else {                           // C region
          int c = (tile - 10) * 16 + n;
#pragma unroll
          for (int v = 0; v < 8; ++v) s_C[(v + rb) * NSTATE + c] = acc[v];
        }
      }
    }
    __syncthreads();

    if (d < DINNER) {
      for (int tt = 0; tt < 16; ++tt) {
        float dl = s_delta[tt * DINNER + d];
        cum += dl;
        const float4* C4 = (const float4*)&s_C[tt * NSTATE];
        float y0 = 0.f, y1 = 0.f, y2 = 0.f, y3 = 0.f;
#pragma unroll
        for (int q = 0; q < 16; ++q) {
          float4 cc = C4[q];
          y0 += hn[4 * q + 0] * fexp2(a2[4 * q + 0] * cum) * cc.x;
          y1 += hn[4 * q + 1] * fexp2(a2[4 * q + 1] * cum) * cc.y;
          y2 += hn[4 * q + 2] * fexp2(a2[4 * q + 2] * cum) * cc.z;
          y3 += hn[4 * q + 3] * fexp2(a2[4 * q + 3] * cum) * cc.w;
        }
        s_y[tt * DINNER + d] = (y0 + y1) + (y2 + y3);
      }
    }
    __syncthreads();
    for (int i = tid; i < 16 * DINNER; i += 128) {
      int r = i / DINNER;
      size_t g = ((size_t)b * SEQ + (tb + r)) * DINNER + (i % DINNER);
      ysk[g] += s_y[i];   // each (b,k,t) owned by exactly one block: no race
    }
    __syncthreads();
  }
}

// ---- kernel 6: direction merge + gating + out_proj (WMMA) -----------------

__global__ void __launch_bounds__(128) outproj_kernel(const float* __restrict__ ys,
                                                      const float* __restrict__ zbuf,
                                                      const float* __restrict__ w,
                                                      float* __restrict__ out) {
  __shared__ __align__(16) float s_y[16 * DINNER];
  int tid = threadIdx.x, lane = tid & 31, wv = tid >> 5;

  bool has2 = (wv < 2);
  v16bf bw0[3], bw1[3];
#pragma unroll
  for (int kc = 0; kc < 3; ++kc) bw0[kc] = b_frag_global(w, DINNER, wv * 16, kc * 32, lane);
  if (has2)
#pragma unroll
    for (int kc = 0; kc < 3; ++kc) bw1[kc] = b_frag_global(w, DINNER, (wv + 4) * 16, kc * 32, lane);

  const float* ys0 = ys;
  const float* ys1 = ys0 + (size_t)BL * DINNER;
  const float* ys2 = ys1 + (size_t)BL * DINNER;
  const float* ys3 = ys2 + (size_t)BL * DINNER;

  for (int ms = 0; ms < 8; ++ms) {
    size_t row0 = ((size_t)blockIdx.x * 8 + ms) * 16;
    for (int i = tid; i < 16 * DINNER; i += 128) {
      size_t r = row0 + i / DINNER; int j = i % DINNER;
      int b = (int)(r / SEQ), l = (int)(r % SEQ);
      size_t fwd = r * DINNER + j;
      size_t rev = ((size_t)b * SEQ + (SEQ - 1 - l)) * DINNER + j;
      float yv = ys0[fwd] + ys2[fwd] + ys1[rev] + ys3[rev];
      s_y[i] = yv * zbuf[fwd];
    }
    __syncthreads();
    v16bf af0 = a_frag_lds(s_y, DINNER, 0, lane, 0);
    v16bf af1 = a_frag_lds(s_y, DINNER, 32, lane, 0);
    v16bf af2 = a_frag_lds(s_y, DINNER, 64, lane, 0);
    {
      v8f acc = {};
      acc = wmma_bf16(af0, bw0[0], acc);
      acc = wmma_bf16(af1, bw0[1], acc);
      acc = wmma_bf16(af2, bw0[2], acc);
      int c = wv * 16 + (lane & 15);
      int rb = (lane >> 4) * 8;
#pragma unroll
      for (int v = 0; v < 8; ++v) out[(row0 + v + rb) * DMODEL + c] = acc[v];
    }
    if (has2) {
      v8f acc = {};
      acc = wmma_bf16(af0, bw1[0], acc);
      acc = wmma_bf16(af1, bw1[1], acc);
      acc = wmma_bf16(af2, bw1[2], acc);
      int c = (wv + 4) * 16 + (lane & 15);
      int rb = (lane >> 4) * 8;
#pragma unroll
      for (int v = 0; v < 8; ++v) out[(row0 + v + rb) * DMODEL + c] = acc[v];
    }
    __syncthreads();
  }
}

// ---- host launch -----------------------------------------------------------

extern "C" void kernel_launch(void* const* d_in, const int* in_sizes, int n_in,
                              void* d_out, int out_size, void* d_ws, size_t ws_size,
                              hipStream_t stream) {
  const float* x    = (const float*)d_in[0];
  const float* ipw  = (const float*)d_in[1];
  const float* cw   = (const float*)d_in[2];
  const float* xpw  = (const float*)d_in[3];
  const float* dtw  = (const float*)d_in[4];
  const float* opw  = (const float*)d_in[5];
  const float* alog = (const float*)d_in[6];
  const float* Dsp  = (const float*)d_in[7];
  float* out = (float*)d_out;

  float* ws = (float*)d_ws;
  size_t uN = (size_t)BL * DINNER;                 // 6,291,456
  float* u      = ws;
  float* zbuf   = u + uN;
  float* xxraw  = zbuf + uN;
  float* ys     = xxraw + uN;                      // 4 * uN
  float* wdelta = ys + 4 * uN;                     // K*96*96
  float* Ssum   = wdelta + (size_t)KDIR * DINNER * DINNER;       // B*K*NC*96
  float* hend   = Ssum + (size_t)BATCH * KDIR * NC * DINNER;     // B*K*NC*96*64

  wdelta_kernel<<<KDIR, 256, 0, stream>>>(dtw, xpw, wdelta);
  inproj_kernel<<<BL / 128, 128, 0, stream>>>(x, ipw, xxraw, zbuf);
  conv_kernel<<<(BL * DINNER) / 256, 256, 0, stream>>>(xxraw, cw, u);
  scan_pass1<<<BATCH * KDIR * NC, 128, 0, stream>>>(u, xpw, wdelta, alog, Dsp, ys, hend, Ssum);
  scan_pass2<<<(BATCH * KDIR * DINNER * NSTATE) / 256, 256, 0, stream>>>(alog, Ssum, hend);
  scan_pass3<<<BATCH * KDIR * (NC - 1), 128, 0, stream>>>(u, xpw, wdelta, alog, hend, ys);
  outproj_kernel<<<BL / 128, 128, 0, stream>>>(ys, zbuf, opw, out);
}
